// SSD_Z_19945828123265
// MI455X (gfx1250) — compile-verified, matrix operand order checked by reading, not run on hardware
//
#include <hip/hip_runtime.h>

typedef __attribute__((ext_vector_type(8)))  __bf16 v8bf;
typedef __attribute__((ext_vector_type(16))) __bf16 v16bf;
typedef __attribute__((ext_vector_type(8)))  float  v8f;
typedef __attribute__((ext_vector_type(4)))  unsigned int v4u;
typedef __attribute__((ext_vector_type(8)))  int v8i;
typedef __attribute__((ext_vector_type(4)))  int v4i;

#define BM 128
#define BN 64
#define KB 32
#define LDK 40   // padded LDS row stride in bf16 (80B) -> conflict-free b128 reads

#if defined(__has_builtin)
# if __has_builtin(__builtin_amdgcn_tensor_load_to_lds) && \
     __has_builtin(__builtin_amdgcn_s_wait_tensorcnt)
#  define HAVE_TDM 1
# endif
#endif

// ---------------------------------------------------------------------------
// Fragment load from LDS: 16 bf16 per lane, matching CDNA5 16-bit A/B layout:
// lanes 0-15 hold K {c0..c0+7, 16+c0..16+c0+7} with c0=0; lanes 16-31 with c0=8.
// Two 16-byte ds_read_b128 per fragment.
// ---------------------------------------------------------------------------
__device__ __forceinline__ v16bf ldfrag(const __bf16* rowp, int c0) {
    v8bf lo = *(const v8bf*)(rowp + c0);
    v8bf hi = *(const v8bf*)(rowp + c0 + 16);
    return __builtin_shufflevector(lo, hi, 0, 1, 2, 3, 4, 5, 6, 7,
                                           8, 9, 10, 11, 12, 13, 14, 15);
}

// ---------------------------------------------------------------------------
// Implicit-GEMM convolution.
//   input : NHWC bf16       [Nb][H][W][Cin]
//   weight: [Cout][K] bf16  with K = KHW*KHW*Cin, k = (kh*KHW+kw)*Cin + ic
//   bias  : f32 [Cout]
//   output: NHWC, bf16 (f32out=0) or f32 (f32out=1), optional ReLU
// Block tile 128x64 (M x N), 8 waves, each wave a 32x32 tile via 4 WMMA accs.
// B (weight) tile is staged into LDS by the Tensor Data Mover when available:
// 2-D tile 32 elem x 64 rows, dim0_stride=K, LDS padding 16dw-interval/4dw-pad
// reproduces the LDK=40 layout; remaining-dim OOB gives zero fill for edges.
// ---------------------------------------------------------------------------
__global__ __launch_bounds__(256)
void conv_igemm_kernel(const __bf16* __restrict__ in, const __bf16* __restrict__ wt,
                       const float* __restrict__ bias, void* __restrict__ outp,
                       int Nb, int H, int W, int Cin, int Hout, int Wout, int Cout,
                       int KHW, int stride, int pad, int relu, int f32out)
{
    __shared__ __bf16 As[BM][LDK];
    __shared__ __bf16 Bs[BN][LDK];

    const int tid  = threadIdx.x;
    const int lane = tid & 31;
    const int wv   = tid >> 5;
    const int wm   = (wv >> 1) << 5;   // 0,32,64,96
    const int wn   = (wv & 1) << 5;    // 0,32

    const int M = Nb * Hout * Wout;
    const int K = KHW * KHW * Cin;
    const int blockM = blockIdx.x * BM;
    const int blockN = blockIdx.y * BN;

    // A staging map: 256 threads cover 128 rows x (2 halves of 16 k-vals)
    const int arow = tid >> 1;
    const int acol = (tid & 1) << 4;   // 0 or 16
    const int m    = blockM + arow;
    const bool mok = m < M;
    int ow = 0, oh = 0, bimg = 0;
    {
        int mm = mok ? m : 0;
        ow = mm % Wout; int t = mm / Wout; oh = t % Hout; bimg = t / Hout;
    }

#ifndef HAVE_TDM
    // B staging map (fallback): 256 threads cover 64 rows x 4 chunks of 8 k-vals
    const int brow = tid >> 2;
    const int bcol = (tid & 3) << 3;
    const int bn   = blockN + brow;
#endif

    const bool fastA = (Cin % KB) == 0;

    v8f acc[2][2] = {};

    for (int k0 = 0; k0 < K; k0 += KB) {
#ifdef HAVE_TDM
        // ---------- stage B tile (64 x 32) via Tensor Data Mover ----------
        if (wv == 0) {
            const unsigned int ldsOff =
                (unsigned int)(unsigned long long)(&Bs[0][0]);  // LDS byte offset
            const unsigned long long ga =
                (unsigned long long)(const void*)(wt + (size_t)blockN * K + k0);
            const unsigned int td0 = (unsigned int)(K - k0);       // remaining k (elems)
            const unsigned int td1 = (unsigned int)(Cout - blockN);// remaining rows
            v4u g0;
            g0.x = 1u;                                     // count=1, load-descriptor
            g0.y = ldsOff;                                 // lds_addr
            g0.z = (unsigned int)ga;                       // global_addr[31:0]
            g0.w = ((unsigned int)(ga >> 32) & 0x01FFFFFFu) | (2u << 30); // hi + type=2
            v8i g1;
            g1[0] = (int)((1u << 16)      // data_size = 1 -> 2-byte elements
                        | (1u << 20)      // pad_enable
                        | (3u << 22)      // pad_interval: 16 dwords (=64B row)
                        | (3u << 25));    // pad_amount : 4 dwords  (=16B -> LDK 80B)
            g1[1] = (int)((td0 & 0xFFFFu) << 16);                        // dim0 lo16
            g1[2] = (int)(((td0 >> 16) & 0xFFFFu) | ((td1 & 0xFFFFu) << 16));
            g1[3] = (int)(((td1 >> 16) & 0xFFFFu) | ((unsigned)KB << 16)); // tile_dim0
            g1[4] = BN;                                    // tile_dim1=64, tile_dim2=0
            g1[5] = K;                                     // tensor_dim0_stride lo32
            g1[6] = 0;
            g1[7] = 0;
            v4i g2 = {0, 0, 0, 0};
            v4i g3 = {0, 0, 0, 0};
#if defined(__clang_major__) && __clang_major__ >= 23
            v8i g4 = {0, 0, 0, 0, 0, 0, 0, 0};
            __builtin_amdgcn_tensor_load_to_lds(g0, g1, g2, g3, g4, 0);
#else
            __builtin_amdgcn_tensor_load_to_lds(g0, g1, g2, g3, 0);
#endif
        }
#endif
        // ------------------ stage A tile (128 x 32) ------------------
        if (fastA) {
            const int plane = k0 / Cin;
            const int kh = plane / KHW, kw = plane - kh * KHW;
            const int icb = (k0 - plane * Cin) + acol;
            const int ih = oh * stride - pad + kh;
            const int iw = ow * stride - pad + kw;
            if (mok && (unsigned)ih < (unsigned)H && (unsigned)iw < (unsigned)W) {
                const __bf16* src = in + ((size_t)(bimg * H + ih) * W + iw) * Cin + icb;
                *(uint4*)&As[arow][acol]     = *(const uint4*)src;
                *(uint4*)&As[arow][acol + 8] = *(const uint4*)(src + 8);
            } else {
                uint4 z = {0u, 0u, 0u, 0u};
                *(uint4*)&As[arow][acol]     = z;
                *(uint4*)&As[arow][acol + 8] = z;
            }
        } else {
            #pragma unroll 4
            for (int j = 0; j < 16; ++j) {
                const int c = acol + j;
                const int k = k0 + c;
                float v = 0.f;
                if (mok && k < K) {
                    const int plane = k / Cin;
                    const int ic = k - plane * Cin;
                    const int kh = plane / KHW, kw = plane - kh * KHW;
                    const int ih = oh * stride - pad + kh;
                    const int iw = ow * stride - pad + kw;
                    if ((unsigned)ih < (unsigned)H && (unsigned)iw < (unsigned)W)
                        v = (float)in[((size_t)(bimg * H + ih) * W + iw) * Cin + ic];
                }
                As[arow][c] = (__bf16)v;
            }
        }
#ifdef HAVE_TDM
        if (wv == 0) __builtin_amdgcn_s_wait_tensorcnt(0);
#else
        // ------------------ stage B tile (64 x 32) fallback ----------
        if (bn < Cout && (K - k0) >= KB) {
            *(uint4*)&Bs[brow][bcol] = *(const uint4*)&wt[(size_t)bn * K + k0 + bcol];
            if (k0 + KB < K)
                __builtin_prefetch(&wt[(size_t)bn * K + k0 + KB + bcol], 0, 1);
        } else {
            #pragma unroll
            for (int j = 0; j < 8; ++j) {
                const int k = k0 + bcol + j;
                Bs[brow][bcol + j] = (bn < Cout && k < K) ? wt[(size_t)bn * K + k]
                                                          : (__bf16)0.f;
            }
        }
#endif
        __syncthreads();

        // ------------------ WMMA compute -----------------------------
        const int c0 = (lane >> 4) << 3;   // 0 for lanes 0-15, 8 for lanes 16-31
        const int la = lane & 15;
        v16bf a0 = ldfrag(&As[wm + la][0], c0);
        v16bf a1 = ldfrag(&As[wm + 16 + la][0], c0);
        v16bf b0 = ldfrag(&Bs[wn + la][0], c0);
        v16bf b1 = ldfrag(&Bs[wn + 16 + la][0], c0);
        acc[0][0] = __builtin_amdgcn_wmma_f32_16x16x32_bf16(false, a0, false, b0,
                        (short)0, acc[0][0], false, false);
        acc[0][1] = __builtin_amdgcn_wmma_f32_16x16x32_bf16(false, a0, false, b1,
                        (short)0, acc[0][1], false, false);
        acc[1][0] = __builtin_amdgcn_wmma_f32_16x16x32_bf16(false, a1, false, b0,
                        (short)0, acc[1][0], false, false);
        acc[1][1] = __builtin_amdgcn_wmma_f32_16x16x32_bf16(false, a1, false, b1,
                        (short)0, acc[1][1], false, false);
        __syncthreads();
    }

    // ------------------ epilogue: bias + relu + store NHWC -----------
    // C/D layout: VGPR r -> lanes 0-15: (M=r, N=lane); lanes 16-31: (M=8+r, N=lane-16)
    const int mrow0 = blockM + wm + ((lane >> 4) << 3);
    const int ncol0 = blockN + wn + (lane & 15);
    float*  outF = (float*)outp;
    __bf16* outB = (__bf16*)outp;
    #pragma unroll
    for (int mt = 0; mt < 2; ++mt) {
        #pragma unroll
        for (int nt = 0; nt < 2; ++nt) {
            const int n = ncol0 + nt * 16;
            if (n >= Cout) continue;
            const float bv = bias[n];
            #pragma unroll
            for (int r = 0; r < 8; ++r) {
                const int mm = mrow0 + mt * 16 + r;
                if (mm >= M) continue;
                float v = acc[mt][nt][r] + bv;
                if (relu) v = fmaxf(v, 0.f);
                const int ow2 = mm % Wout;
                int t = mm / Wout;
                const int oh2 = t % Hout;
                const int b2  = t / Hout;
                const size_t oidx = ((size_t)(b2 * Hout + oh2) * Wout + ow2) * Cout + n;
                if (f32out) outF[oidx] = v;
                else        outB[oidx] = (__bf16)v;
            }
        }
    }
}

// ---------------------------------------------------------------------------
// OIHW f32  ->  [O][(kh*KHW+kw)*Cin+ic] bf16
// ---------------------------------------------------------------------------
__global__ void convert_w_kernel(const float* __restrict__ w, __bf16* __restrict__ wt,
                                 int O, int Cin, int KHW)
{
    const int K = KHW * KHW * Cin;
    const size_t total = (size_t)O * K;
    const size_t idx = (size_t)blockIdx.x * blockDim.x + threadIdx.x;
    if (idx >= total) return;
    const int o  = (int)(idx / K);
    const int kk = (int)(idx - (size_t)o * K);
    const int plane = kk / Cin;
    const int ic = kk - plane * Cin;
    const int kh = plane / KHW, kw = plane - kh * KHW;
    wt[idx] = (__bf16)w[(((size_t)o * Cin + ic) * KHW + kh) * KHW + kw];
}

// ---------------------------------------------------------------------------
// NCHW f32 -> NHWC bf16
// ---------------------------------------------------------------------------
__global__ void nchw2nhwc_kernel(const float* __restrict__ x, __bf16* __restrict__ y,
                                 int N, int C, int H, int W)
{
    const size_t total = (size_t)N * C * H * W;
    const size_t idx = (size_t)blockIdx.x * blockDim.x + threadIdx.x;
    if (idx >= total) return;
    const int c = (int)(idx % C);
    size_t t = idx / C;
    const int wpos = (int)(t % W); t /= W;
    const int h = (int)(t % H);
    const int n = (int)(t / H);
    y[idx] = (__bf16)x[(((size_t)n * C + c) * H + h) * W + wpos];
}

// ---------------------------------------------------------------------------
// L2Norm over channels with learned gain: one wave32 per pixel
// ---------------------------------------------------------------------------
__global__ void l2norm_kernel(const __bf16* __restrict__ in, const float* __restrict__ gain,
                              __bf16* __restrict__ out, int npix, int C)
{
    const int gid  = blockIdx.x * blockDim.x + threadIdx.x;
    const int wave = gid >> 5;
    const int lane = gid & 31;
    if (wave >= npix) return;
    const __bf16* p = in + (size_t)wave * C;
    float s = 0.f;
    for (int c = lane; c < C; c += 32) { float v = (float)p[c]; s += v * v; }
    #pragma unroll
    for (int off = 16; off; off >>= 1) s += __shfl_xor(s, off, 32);
    const float inv = 1.0f / (sqrtf(s) + 1e-10f);
    __bf16* q = out + (size_t)wave * C;
    for (int c = lane; c < C; c += 32) q[c] = (__bf16)((float)p[c] * inv * gain[c]);
}

// ---------------------------------------------------------------------------
// Gather (permute/concat/reshape of reference) + 2-way softmax -> f32 output
// out shape (8, 28, 21824, 2). pair index: per-source (y*W + x)*4 + anchor.
// head-conv channel = head*8 + anchor*2 + p, stored NHWC f32 per source.
// ---------------------------------------------------------------------------
__global__ void softmax_kernel(const float* __restrict__ h0, const float* __restrict__ h1,
                               const float* __restrict__ h2, const float* __restrict__ h3,
                               const float* __restrict__ h4, float* __restrict__ out)
{
    const int TOT = 21824;
    const size_t idx = (size_t)blockIdx.x * blockDim.x + threadIdx.x;
    if (idx >= (size_t)8 * 28 * TOT) return;
    const int pair = (int)(idx % TOT);
    const int t    = (int)(idx / TOT);
    const int head = t % 28;
    const int b    = t / 28;
    const float* hp; int HW; int pl;
    if      (pair < 16384) { hp = h0; HW = 4096; pl = pair;         }
    else if (pair < 20480) { hp = h1; HW = 1024; pl = pair - 16384; }
    else if (pair < 21504) { hp = h2; HW = 256;  pl = pair - 20480; }
    else if (pair < 21760) { hp = h3; HW = 64;   pl = pair - 21504; }
    else                   { hp = h4; HW = 16;   pl = pair - 21760; }
    const int a   = pl & 3;
    const int pix = pl >> 2;
    const size_t base = ((size_t)b * HW + pix) * 224 + head * 8 + a * 2;
    const float v0 = hp[base], v1 = hp[base + 1];
    const float mx = fmaxf(v0, v1);
    const float e0 = __expf(v0 - mx), e1 = __expf(v1 - mx);
    const float inv = 1.f / (e0 + e1);
    out[idx * 2]     = e0 * inv;
    out[idx * 2 + 1] = e1 * inv;
}

// ---------------------------------------------------------------------------
// Host orchestration
// ---------------------------------------------------------------------------
extern "C" void kernel_launch(void* const* d_in, const int* in_sizes, int n_in,
                              void* d_out, int out_size, void* d_ws, size_t ws_size,
                              hipStream_t stream)
{
    (void)in_sizes; (void)n_in; (void)out_size; (void)ws_size;

    const float* x   = (const float*)d_in[0];
    const float* l2g = (const float*)d_in[1];
    const float* bw[6]; const float* bbp[6];
    const float* ewp[6]; const float* ebp[6];
    const float* hwp[5]; const float* hbp[5];
    for (int i = 0; i < 6; ++i) { bw[i]  = (const float*)d_in[2 + i];  bbp[i] = (const float*)d_in[8 + i]; }
    for (int i = 0; i < 6; ++i) { ewp[i] = (const float*)d_in[14 + i]; ebp[i] = (const float*)d_in[20 + i]; }
    for (int i = 0; i < 5; ++i) { hwp[i] = (const float*)d_in[26 + i]; hbp[i] = (const float*)d_in[31 + i]; }

    // geometry
    const int bci[6] = {3, 64, 128, 256, 512, 512};
    const int bco[6] = {64, 128, 256, 512, 512, 1024};
    const int eci[6] = {1024, 256, 512, 128, 256, 128};
    const int eco[6] = {256, 512, 128, 256, 128, 256};
    const int ekk[6] = {1, 3, 1, 3, 1, 3};
    const int scc[5] = {512, 1024, 512, 256, 256};
    const int shw[5] = {64, 32, 16, 8, 4};

    // bump allocator on workspace
    char* ws = (char*)d_ws;
    size_t off = 0;
    auto alloc = [&](size_t bytes) -> void* {
        void* p = ws + off;
        off = (off + bytes + 255) & ~(size_t)255;
        return p;
    };
    auto cdiv = [](size_t a, size_t b) { return (a + b - 1) / b; };

    __bf16* wtb[6]; __bf16* wte[6]; __bf16* wth[5];
    for (int i = 0; i < 6; ++i) wtb[i] = (__bf16*)alloc((size_t)bco[i] * bci[i] * 9 * sizeof(__bf16));
    for (int i = 0; i < 6; ++i) wte[i] = (__bf16*)alloc((size_t)eco[i] * eci[i] * ekk[i] * ekk[i] * sizeof(__bf16));
    for (int i = 0; i < 5; ++i) wth[i] = (__bf16*)alloc((size_t)224 * scc[i] * 9 * sizeof(__bf16));

    __bf16* xin  = (__bf16*)alloc((size_t)8 * 512 * 512 * 3 * sizeof(__bf16));
    __bf16* bufA = (__bf16*)alloc((size_t)8 * 512 * 512 * 64 * sizeof(__bf16));   // max odd-layer act
    __bf16* bufB = (__bf16*)alloc((size_t)8 * 256 * 256 * 128 * sizeof(__bf16));  // max even-layer act
    __bf16* srcb[5];
    for (int i = 0; i < 5; ++i) srcb[i] = (__bf16*)alloc((size_t)8 * shw[i] * shw[i] * scc[i] * sizeof(__bf16));
    float* hout[5];
    for (int i = 0; i < 5; ++i) hout[i] = (float*)alloc((size_t)8 * shw[i] * shw[i] * 224 * sizeof(float));

    // weight + input conversions
    for (int i = 0; i < 6; ++i) {
        size_t tot = (size_t)bco[i] * bci[i] * 9;
        convert_w_kernel<<<dim3((unsigned)cdiv(tot, 256)), 256, 0, stream>>>(bw[i], wtb[i], bco[i], bci[i], 3);
    }
    for (int i = 0; i < 6; ++i) {
        size_t tot = (size_t)eco[i] * eci[i] * ekk[i] * ekk[i];
        convert_w_kernel<<<dim3((unsigned)cdiv(tot, 256)), 256, 0, stream>>>(ewp[i], wte[i], eco[i], eci[i], ekk[i]);
    }
    for (int i = 0; i < 5; ++i) {
        size_t tot = (size_t)224 * scc[i] * 9;
        convert_w_kernel<<<dim3((unsigned)cdiv(tot, 256)), 256, 0, stream>>>(hwp[i], wth[i], 224, scc[i], 3);
    }
    {
        size_t tot = (size_t)8 * 3 * 512 * 512;
        nchw2nhwc_kernel<<<dim3((unsigned)cdiv(tot, 256)), 256, 0, stream>>>(x, xin, 8, 3, 512, 512);
    }

    int H = 512, W = 512;
    auto runconv = [&](const __bf16* inp, const __bf16* wt, const float* bias, void* outp,
                       int Ci, int Co, int kk, int st, int pd, int relu, int f32o) {
        const int Ho = (H + 2 * pd - kk) / st + 1;
        const int Wo = (W + 2 * pd - kk) / st + 1;
        const size_t M = (size_t)8 * Ho * Wo;
        dim3 grid((unsigned)cdiv(M, BM), (unsigned)cdiv((size_t)Co, BN));
        conv_igemm_kernel<<<grid, 256, 0, stream>>>(inp, wt, bias, outp,
                                                    8, H, W, Ci, Ho, Wo, Co,
                                                    kk, st, pd, relu, f32o);
        H = Ho; W = Wo;
    };

    // backbone
    runconv(xin,  wtb[0], bbp[0], bufA, 3,    64,  3, 1, 1, 1, 0);   // 512x512x64
    runconv(bufA, wtb[1], bbp[1], bufB, 64,   128, 3, 2, 1, 1, 0);   // 256x256x128
    runconv(bufB, wtb[2], bbp[2], bufA, 128,  256, 3, 2, 1, 1, 0);   // 128x128x256
    runconv(bufA, wtb[3], bbp[3], bufB, 256,  512, 3, 2, 1, 1, 0);   // 64x64x512
    {   // source 0: L2Norm(conv4 out) * gain
        const int npix = 8 * H * W;   // 8*64*64
        const size_t thr = (size_t)npix * 32;
        l2norm_kernel<<<dim3((unsigned)cdiv(thr, 256)), 256, 0, stream>>>(bufB, l2g, srcb[0], npix, 512);
    }
    runconv(bufB, wtb[4], bbp[4], bufA,    512, 512,  3, 2, 1, 1, 0); // 32x32x512
    runconv(bufA, wtb[5], bbp[5], srcb[1], 512, 1024, 3, 1, 1, 1, 0); // 32x32x1024  (source 1)

    // extras (alternating 1x1 s1 p0 / 3x3 s2 p1; source after each odd conv)
    runconv(srcb[1], wte[0], ebp[0], bufA,    1024, 256, 1, 1, 0, 1, 0); // 32x32x256
    runconv(bufA,    wte[1], ebp[1], srcb[2], 256,  512, 3, 2, 1, 1, 0); // 16x16x512  (source 2)
    runconv(srcb[2], wte[2], ebp[2], bufA,    512,  128, 1, 1, 0, 1, 0); // 16x16x128
    runconv(bufA,    wte[3], ebp[3], srcb[3], 128,  256, 3, 2, 1, 1, 0); // 8x8x256    (source 3)
    runconv(srcb[3], wte[4], ebp[4], bufA,    256,  128, 1, 1, 0, 1, 0); // 8x8x128
    runconv(bufA,    wte[5], ebp[5], srcb[4], 128,  256, 3, 2, 1, 1, 0); // 4x4x256    (source 4)

    // conf heads: 3x3 s1 p1, no relu, f32 output for softmax
    for (int i = 0; i < 5; ++i) {
        H = shw[i]; W = shw[i];
        runconv(srcb[i], wth[i], hbp[i], hout[i], scc[i], 224, 3, 1, 1, 0, 1);
    }

    // final gather + pairwise softmax into d_out (8, 28, 21824, 2) f32
    {
        const size_t tot = (size_t)8 * 28 * 21824;
        softmax_kernel<<<dim3((unsigned)cdiv(tot, 256)), 256, 0, stream>>>(
            hout[0], hout[1], hout[2], hout[3], hout[4], (float*)d_out);
    }
}
